// mha0_13795434955003
// MI455X (gfx1250) — compile-verified, matrix operand order checked by reading, not run on hardware
//
#include <hip/hip_runtime.h>
#include <hip/hip_bf16.h>
#include <math.h>

// ---------------------------------------------------------------------------
// MHA with softmax-over-heads, fused for MI455X (gfx1250, wave32, WMMA).
//   proj    : KEY/QUERY @ W^T + b -> f16 head-split; VALUE @ Wv^T + b -> f16
//             head-split TRANSPOSED (PvT[b][h][d][L]) so the attention kernel
//             reads every operand contiguously.
//   attn    : fused score + head-softmax + attn^T@Wv, K tiles staged to LDS
//             with global_load_async_to_lds_b128 (double-buffered, ASYNCcnt).
//   outproj : ctx @ Wo^T + b -> f32
// All matmuls via v_wmma_f32_16x16x32_f16 (f32 accumulate).
// ---------------------------------------------------------------------------

typedef __attribute__((ext_vector_type(16))) _Float16 v16h;
typedef __attribute__((ext_vector_type(8)))  float    v8f;

#define DEV static __device__ __forceinline__

constexpr int cB = 4, cL = 2048, cINK = 256, cHID = 1024, cHEADS = 8, cD = 128, cOUT = 256;
constexpr float cSCALE = 0.08838834764831845f;   // 1/sqrt(128)

DEV int lane_id() { return (int)(threadIdx.x & 31u); }

// K index (within the 32-wide K tile) held by element e of a 16-bit A/B frag.
// Per ISA 7.12.2: lanes 0-15 -> K 0..7 / 16..23, lanes 16-31 -> K 8..15 / 24..31.
DEV int kidx(int lane, int e) {
  int kb = (lane & 16) ? 8 : 0;
  return (e < 8) ? (kb + e) : (16 + kb + (e - 8));
}

// A/B fragment from a row-major f32 tile (row/col = lane&15, K along elements)
DEV v16h frag_rm_f32(const float* __restrict__ base, int ld) {
  int lane = lane_id();
  const float* p = base + (size_t)(lane & 15) * ld;
  v16h f;
#pragma unroll
  for (int e = 0; e < 16; ++e) f[e] = (_Float16)p[kidx(lane, e)];
  return f;
}

// A/B fragment from a row-major f16 tile (two contiguous 16B chunks per lane)
DEV v16h frag_rm_f16(const _Float16* base, int ld) {
  int lane = lane_id();
  const _Float16* p = base + (size_t)(lane & 15) * ld;
  v16h f;
#pragma unroll
  for (int e = 0; e < 16; ++e) f[e] = p[kidx(lane, e)];
  return f;
}

DEV v8f wmma16(v16h a, v16h b, v8f c) {
  return __builtin_amdgcn_wmma_f32_16x16x32_f16(false, a, false, b, (short)0, c,
                                                false, false);
}

// Async-copy 512B (32 lanes x 16B) from global to LDS; CDNA5 async path.
// lds_off is the byte offset within the workgroup's LDS segment.
DEV void async_copy_512(unsigned lds_off, const void* gptr) {
  asm volatile("global_load_async_to_lds_b128 %0, %1, off"
               :: "v"(lds_off), "v"(gptr) : "memory");
}
DEV void wait_async_le(int) {}  // (documentation only; real waits below)
DEV void wait_async16() { asm volatile("s_wait_asynccnt 0x10" ::: "memory"); }
DEV void wait_async0()  { asm volatile("s_wait_asynccnt 0x0"  ::: "memory"); }

// ---------------------------------------------------------------------------
// Kernel A: projections.  blockIdx.y: 0->Pk(KEY,W), 1->Pq(QUERY,W), 2->PvT(VALUE,Wv)
// One wave per 16x16 output tile; K=256 -> 8 WMMAs.
// Pk/Pq layout: P[b][h][l][d];  PvT layout: P[b][h][d][l]  (h = hidden%8, d = hidden/8)
// ---------------------------------------------------------------------------
__global__ void proj_kernel(const float* __restrict__ KEY,
                            const float* __restrict__ VALUE,
                            const float* __restrict__ QUERY,
                            const float* __restrict__ W_w,
                            const float* __restrict__ W_b,
                            const float* __restrict__ Wv_w,
                            const float* __restrict__ Wv_b,
                            _Float16* __restrict__ Pk,
                            _Float16* __restrict__ Pq,
                            _Float16* __restrict__ PvT) {
  const int which = blockIdx.y;
  const float* src  = (which == 0) ? KEY : (which == 1) ? QUERY : VALUE;
  const float* w    = (which == 2) ? Wv_w : W_w;
  const float* bias = (which == 2) ? Wv_b : W_b;
  _Float16*    dst  = (which == 0) ? Pk : (which == 1) ? Pq : PvT;

  const int NT = cHID / 16;  // 64
  const int tile = blockIdx.x * 4 + (threadIdx.x >> 5);
  const int mt = tile / NT, nt = tile % NT;
  const int lane = lane_id();

  v8f acc = {};
#pragma unroll
  for (int k0 = 0; k0 < cINK; k0 += 32) {
    v16h a = frag_rm_f32(src + (size_t)(mt * 16) * cINK + k0, cINK);
    v16h b = frag_rm_f32(w   + (size_t)(nt * 16) * cINK + k0, cINK);
    acc = wmma16(a, b, acc);
  }

  const int n = nt * 16 + (lane & 15);
  const float bv = bias[n];
  const int h = n & 7, d = n >> 3;
#pragma unroll
  for (int r = 0; r < 8; ++r) {
    int m = mt * 16 + r + ((lane & 16) ? 8 : 0);
    int bi = m / cL, l = m % cL;
    _Float16 val = (_Float16)(acc[r] + bv);
    if (which == 2)
      dst[(((size_t)bi * cHEADS + h) * cD + d) * cL + l] = val;   // transposed
    else
      dst[(((size_t)bi * cHEADS + h) * cL + l) * cD + d] = val;
  }
}

// ---------------------------------------------------------------------------
// Kernel B: fused attention. One block per (b, 16-col Lq tile); 8 waves = 8 heads.
// K tiles double-buffered in LDS via async copies; Q fragments hoisted; score
// WMMAs -> LDS -> cross-head softmax (elementwise in (i,k)) -> PvT@attn WMMAs.
// ---------------------------------------------------------------------------
struct AttnSmem {
  _Float16 sK[2][cHEADS][32][cD];   // 128 KB, double-buffered K tiles (first!)
  float    s_sc[cHEADS][32][16];    // 16 KB score/attn exchange
};

__global__ void attn_kernel(const _Float16* __restrict__ Pk,
                            const _Float16* __restrict__ Pq,
                            const _Float16* __restrict__ PvT,
                            _Float16* __restrict__ ctx) {
  __shared__ AttnSmem sm;

  const int bb = blockIdx.x >> 7;          // / (L/16 = 128)
  const int kt = blockIdx.x & 127;
  const int h  = threadIdx.x >> 5;         // wave id == head
  const int lane = lane_id();

  const _Float16* Kh = Pk  + ((size_t)(bb * cHEADS + h) * cL) * cD;
  const _Float16* Qt = Pq  + ((size_t)(bb * cHEADS + h) * cL) * cD
                           + (size_t)(kt * 16) * cD;
  const _Float16* Vt = PvT + ((size_t)(bb * cHEADS + h) * cD) * cL;  // [d][L]

  // Q fragments are invariant over the whole Lk loop: hoist (4 x v16h).
  v16h bq[4];
#pragma unroll
  for (int q = 0; q < 4; ++q) bq[q] = frag_rm_f16(Qt + q * 32, cD);

  v8f acc[8] = {};                         // 8 j-tiles (d = 0..127) x 16 k cols

  // --- async stage of one 32x128 f16 K tile (8 KB) into sK[buf][h] --------
  auto stage_k = [&](int buf, int i0) {
    unsigned base = (unsigned)((buf * cHEADS + h) * (32 * cD * 2));  // sK @ LDS off 0
    const char* g = (const char*)(Kh + (size_t)i0 * cD);
#pragma unroll
    for (int t = 0; t < 16; ++t) {
      unsigned lds = base + (unsigned)(t * 512 + lane * 16);
      async_copy_512(lds, g + t * 512 + lane * 16);
    }
  };

  stage_k(0, 0);

  for (int i0 = 0; i0 < cL; i0 += 32) {
    const int  cur  = (i0 >> 5) & 1;
    const bool more = (i0 + 32) < cL;
    if (more) stage_k(cur ^ 1, i0 + 32);

    // prefetch next PvT tile rows into cache while we compute
    if (more) {
#pragma unroll
      for (int r = 0; r < 4; ++r)
        __builtin_prefetch(Vt + (size_t)(r * 32 + lane) * cL + i0 + 32, 0, 1);
    }

    if (more) wait_async16();   // oldest 16 (current buffer) complete
    else      wait_async0();

    const _Float16* sk = &sm.sK[cur][h][0][0];

    // --- score tiles S[i0..i0+31][kt*16..+16] for this head (from LDS) ---
    v8f s0 = {}, s1 = {};
#pragma unroll
    for (int k0 = 0; k0 < cD; k0 += 32) {
      v16h a0 = frag_rm_f16(sk + k0, cD);
      v16h a1 = frag_rm_f16(sk + 16 * cD + k0, cD);
      s0 = wmma16(a0, bq[k0 >> 5], s0);
      s1 = wmma16(a1, bq[k0 >> 5], s1);
    }
    __syncthreads();   // prior iteration's s_sc reads are done
    {
      int kk = lane & 15;
      int mb = (lane & 16) ? 8 : 0;
#pragma unroll
      for (int r = 0; r < 8; ++r) {
        sm.s_sc[h][r + mb][kk]      = s0[r];
        sm.s_sc[h][16 + r + mb][kk] = s1[r];
      }
    }
    __syncthreads();
    // --- softmax across the 8 heads, elementwise in (i,k) ----------------
    for (int t = threadIdx.x; t < 512; t += 256) {
      int ii = t >> 4, kk = t & 15;
      float v[cHEADS];
      float m = -3.0e38f;
#pragma unroll
      for (int hh = 0; hh < cHEADS; ++hh) {
        v[hh] = sm.s_sc[hh][ii][kk] * cSCALE;
        m = fmaxf(m, v[hh]);
      }
      float sum = 0.f;
#pragma unroll
      for (int hh = 0; hh < cHEADS; ++hh) {
        float e = __expf(v[hh] - m);
        v[hh] = e;
        sum += e;
      }
      float inv = 1.f / sum;
#pragma unroll
      for (int hh = 0; hh < cHEADS; ++hh) sm.s_sc[hh][ii][kk] = v[hh] * inv;
    }
    __syncthreads();
    // --- out_h[j,k] += PvT[j,i] @ attn[i,k]  (K = 32 rows of i) ----------
    v16h battn;
    {
      int kk = lane & 15;
      const float* ab = &sm.s_sc[h][0][0];
#pragma unroll
      for (int e = 0; e < 16; ++e)
        battn[e] = (_Float16)ab[kidx(lane, e) * 16 + kk];
    }
#pragma unroll
    for (int jt = 0; jt < 8; ++jt) {
      v16h av = frag_rm_f16(Vt + (size_t)(jt * 16) * cL + i0, cL);
      acc[jt] = wmma16(av, battn, acc[jt]);
    }
  }

  // --- store ctx[b][l][d*8 + h] (f16), the reference's head interleave ----
  const int kk = lane & 15;
  const int l = kt * 16 + kk;
  const int mb = (lane & 16) ? 8 : 0;
#pragma unroll
  for (int jt = 0; jt < 8; ++jt) {
#pragma unroll
    for (int r = 0; r < 8; ++r) {
      int d = jt * 16 + r + mb;
      ctx[((size_t)bb * cL + l) * cHID + d * cHEADS + h] = (_Float16)acc[jt][r];
    }
  }
}

// ---------------------------------------------------------------------------
// Kernel C: final projection ctx[8192,1024](f16) @ Wo^T[1024,256] + b -> f32.
// One wave per 16x16 tile; K=1024 -> 32 WMMAs.
// ---------------------------------------------------------------------------
__global__ void outproj_kernel(const _Float16* __restrict__ ctx,
                               const float* __restrict__ Wo_w,
                               const float* __restrict__ Wo_b,
                               float* __restrict__ out) {
  const int NT = cOUT / 16;  // 16
  const int tile = blockIdx.x * 4 + (threadIdx.x >> 5);
  const int mt = tile / NT, nt = tile % NT;
  const int lane = lane_id();

  v8f acc = {};
  for (int k0 = 0; k0 < cHID; k0 += 32) {
    v16h a = frag_rm_f16(ctx  + (size_t)(mt * 16) * cHID + k0, cHID);
    v16h b = frag_rm_f32(Wo_w + (size_t)(nt * 16) * cHID + k0, cHID);
    acc = wmma16(a, b, acc);
  }

  const int n = nt * 16 + (lane & 15);
  const float bv = Wo_b[n];
#pragma unroll
  for (int r = 0; r < 8; ++r) {
    int m = mt * 16 + r + ((lane & 16) ? 8 : 0);
    out[(size_t)m * cOUT + n] = acc[r] + bv;
  }
}

// ---------------------------------------------------------------------------
extern "C" void kernel_launch(void* const* d_in, const int* in_sizes, int n_in,
                              void* d_out, int out_size, void* d_ws, size_t ws_size,
                              hipStream_t stream) {
  (void)in_sizes; (void)n_in; (void)out_size; (void)ws_size;
  const float* KEY   = (const float*)d_in[0];
  const float* VALUE = (const float*)d_in[1];
  const float* QUERY = (const float*)d_in[2];
  const float* W_w   = (const float*)d_in[3];
  const float* W_b   = (const float*)d_in[4];
  const float* Wv_w  = (const float*)d_in[5];
  const float* Wv_b  = (const float*)d_in[6];
  const float* Wo_w  = (const float*)d_in[7];
  const float* Wo_b  = (const float*)d_in[8];
  float* out = (float*)d_out;

  // f16 workspace: Pk, Pq, PvT (head-split) + ctx. 4 x 16 MB = 64 MB.
  const size_t pelems = (size_t)cB * cHEADS * cL * cD;   // == B*L*HID
  _Float16* Pk  = (_Float16*)d_ws;
  _Float16* Pq  = Pk + pelems;
  _Float16* PvT = Pq + pelems;
  _Float16* ctx = PvT + pelems;

  {
    const int MT = (cB * cL) / 16;           // 512
    const int NT = cHID / 16;                // 64
    dim3 grid((MT * NT) / 4, 3);             // 4 waves/block, 3 projections
    proj_kernel<<<grid, 128, 0, stream>>>(KEY, VALUE, QUERY, W_w, W_b, Wv_w, Wv_b,
                                          Pk, Pq, PvT);
  }
  {
    dim3 grid(cB * (cL / 16));               // 512 blocks, 8 waves = 8 heads
    attn_kernel<<<grid, 256, 0, stream>>>(Pk, Pq, PvT, ctx);
  }
  {
    const int MT = (cB * cL) / 16;           // 512
    const int NT = cOUT / 16;                // 16
    dim3 grid((MT * NT) / 4);                // 2048
    outproj_kernel<<<grid, 128, 0, stream>>>(ctx, Wo_w, Wo_b, out);
  }
}
